// QMixtralAttention_30820685316559
// MI455X (gfx1250) — 16-bit and 8-bit matrix operand order repaired and checked by reading, compiled, not run on hardware
//
#include <hip/hip_runtime.h>
#include <hip/hip_bf16.h>

#define Hdim 4096
#define Tn   2048
#define NHq  32
#define HDm  128
#define NKVh 8
#define KVD  1024

typedef __attribute__((ext_vector_type(16))) _Float16 v16h;
typedef __attribute__((ext_vector_type(8)))  float    v8f;
typedef __attribute__((ext_vector_type(8)))  int      v8i;

__global__ void rmsnorm_quant(const float* __restrict__ x, const float* __restrict__ w,
                              signed char* __restrict__ xq, float* __restrict__ sx) {
  __shared__ float red[256];
  const int t = blockIdx.x;
  const float* row = x + (size_t)t * Hdim;

  float ss = 0.f;
  for (int i = threadIdx.x; i < Hdim; i += 256) { float v = row[i]; ss += v * v; }
  red[threadIdx.x] = ss; __syncthreads();
  for (int st = 128; st > 0; st >>= 1) {
    if (threadIdx.x < st) red[threadIdx.x] += red[threadIdx.x + st];
    __syncthreads();
  }
  const float rinv = rsqrtf(red[0] / (float)Hdim + 1e-5f);
  __syncthreads();

  float amax = 0.f;
  for (int i = threadIdx.x; i < Hdim; i += 256) {
    float y = row[i] * rinv * w[i];
    amax = fmaxf(amax, fabsf(y));
  }
  red[threadIdx.x] = amax; __syncthreads();
  for (int st = 128; st > 0; st >>= 1) {
    if (threadIdx.x < st) red[threadIdx.x] = fmaxf(red[threadIdx.x], red[threadIdx.x + st]);
    __syncthreads();
  }
  const float scale = fmaxf(red[0] / 127.f, 1e-8f);
  if (threadIdx.x == 0) sx[t] = scale;
  const float inv = 1.f / scale;
  for (int i = threadIdx.x; i < Hdim; i += 256) {
    float y = row[i] * rinv * w[i];
    int q = (int)rintf(y * inv);
    q = q < -127 ? -127 : (q > 127 ? 127 : q);
    xq[(size_t)t * Hdim + i] = (signed char)q;
  }
}

__global__ void weight_quant(const float* __restrict__ W, signed char* __restrict__ qw,
                             float* __restrict__ sw) {
  __shared__ float red[256];
  const int o = blockIdx.x;
  const float* row = W + (size_t)o * Hdim;
  float amax = 0.f;
  for (int i = threadIdx.x; i < Hdim; i += 256) amax = fmaxf(amax, fabsf(row[i]));
  red[threadIdx.x] = amax; __syncthreads();
  for (int st = 128; st > 0; st >>= 1) {
    if (threadIdx.x < st) red[threadIdx.x] = fmaxf(red[threadIdx.x], red[threadIdx.x + st]);
    __syncthreads();
  }
  const float scale = fmaxf(red[0] / 127.f, 1e-8f);
  if (threadIdx.x == 0) sw[o] = scale;
  const float inv = 1.f / scale;
  for (int i = threadIdx.x; i < Hdim; i += 256) {
    int q = (int)rintf(row[i] * inv);
    q = q < -127 ? -127 : (q > 127 ? 127 : q);
    qw[(size_t)o * Hdim + i] = (signed char)q;
  }
}

__global__ void gemm_i8_wmma(const signed char* __restrict__ Xq,
                             const signed char* __restrict__ Wq,
                             const float* __restrict__ sx,
                             const float* __restrict__ sw,
                             float* __restrict__ Y, int N) {
  const int lane = threadIdx.x & 31;
  const int wid  = threadIdx.x >> 5;
  const int m0 = blockIdx.y * 32 + (wid >> 1) * 16;
  const int n0 = blockIdx.x * 64 + (wid & 1) * 32;
  const int mr = lane & 15, hi = lane >> 4;

  const signed char* arow  = Xq + (size_t)(m0 + mr) * Hdim;
  const signed char* brow0 = Wq + (size_t)(n0 + mr) * Hdim;
  const signed char* brow1 = Wq + (size_t)(n0 + 16 + mr) * Hdim;

  v8i acc0 = {}, acc1 = {};
  for (int k0 = 0; k0 < Hdim; k0 += 64) {
    __builtin_prefetch(arow + k0 + 512, 0, 1);
    __builtin_prefetch(brow0 + k0 + 512, 0, 1);
    __builtin_prefetch(brow1 + k0 + 512, 0, 1);
    v8i a, b0, b1;
    int* ai  = (int*)&a;
    int* b0i = (int*)&b0;
    int* b1i = (int*)&b1;
#pragma unroll
    for (int d = 0; d < 8; d++) {
      const int ka = 32 * (d >> 2) + hi * 16 + (d & 3) * 4;
      ai[d] = *(const int*)(arow + k0 + ka);
      const int kb = 32 * (d >> 2) + hi * 16 + (d & 3) * 4;
      b0i[d] = *(const int*)(brow0 + k0 + kb);
      b1i[d] = *(const int*)(brow1 + k0 + kb);
    }
    acc0 = __builtin_amdgcn_wmma_i32_16x16x64_iu8(true, a, true, b0, acc0, false, false);
    acc1 = __builtin_amdgcn_wmma_i32_16x16x64_iu8(true, a, true, b1, acc1, false, false);
  }
#pragma unroll
  for (int r = 0; r < 8; r++) {
    const int row = m0 + r + hi * 8;
    const float srow = sx[row];
    const int c0 = n0 + mr;
    const int c1 = n0 + 16 + mr;
    Y[(size_t)row * N + c0] = (float)acc0[r] * srow * sw[c0];
    Y[(size_t)row * N + c1] = (float)acc1[r] * srow * sw[c1];
  }
}

__global__ void quant4_group(float* __restrict__ X, int ngroups) {
  const int warp = (int)((blockIdx.x * blockDim.x + threadIdx.x) >> 5);
  if (warp >= ngroups) return;
  const int lane = threadIdx.x & 31;
  float* p = X + (size_t)warp * 128 + lane * 4;
  float4 v = *(float4*)p;
  float mn = fminf(fminf(v.x, v.y), fminf(v.z, v.w));
  float mx = fmaxf(fmaxf(v.x, v.y), fmaxf(v.z, v.w));
#pragma unroll
  for (int m = 16; m > 0; m >>= 1) {
    mn = fminf(mn, __shfl_xor(mn, m, 32));
    mx = fmaxf(mx, __shfl_xor(mx, m, 32));
  }
  const float scale = fmaxf((mx - mn) / 15.f, 1e-8f);
  const float inv = 1.f / scale;
  v.x = fminf(fmaxf(rintf((v.x - mn) * inv), 0.f), 15.f) * scale + mn;
  v.y = fminf(fmaxf(rintf((v.y - mn) * inv), 0.f), 15.f) * scale + mn;
  v.z = fminf(fmaxf(rintf((v.z - mn) * inv), 0.f), 15.f) * scale + mn;
  v.w = fminf(fmaxf(rintf((v.w - mn) * inv), 0.f), 15.f) * scale + mn;
  *(float4*)p = v;
}

__global__ void rope_to_f16(const float* __restrict__ Y, const int* __restrict__ pos,
                            _Float16* __restrict__ out, int nheads) {
  const int t = blockIdx.x, h = blockIdx.y, d = threadIdx.x;
  const size_t base = (size_t)t * nheads * HDm + (size_t)h * HDm;
  const float xv = Y[base + d];
  const float other = (d < 64) ? -Y[base + d + 64] : Y[base + d - 64];
  const float p = (float)pos[t];
  const float invf = powf(1.0e6f, -(float)(d & 63) / 64.f);
  const float ang = p * invf;
  const float r = xv * cosf(ang) + other * sinf(ang);
  out[((size_t)h * Tn + t) * HDm + d] = (_Float16)r;
}

__global__ void v_to_f16_t(const float* __restrict__ Yv, _Float16* __restrict__ Vt) {
  const int t = blockIdx.x, kv = blockIdx.y, d = threadIdx.x;
  const float v = Yv[(size_t)t * KVD + (size_t)kv * HDm + d];
  Vt[((size_t)kv * HDm + d) * Tn + t] = (_Float16)v;
}

__global__ void attn_fmha(const _Float16* __restrict__ Qh, const _Float16* __restrict__ Kh,
                          const _Float16* __restrict__ Vt, float* __restrict__ Ao) {
  __shared__ _Float16 pbuf[16][32];
  __shared__ float rowm[16], rowl[16];

  const int lane = threadIdx.x;
  const int h = blockIdx.y, kvh = h >> 2;
  const int q0 = blockIdx.x * 16;
  const int mr = lane & 15, hi = lane >> 4;

  const _Float16* Qp = Qh + ((size_t)h * Tn + q0) * HDm;
  const _Float16* Kp = Kh + (size_t)kvh * Tn * HDm;
  const _Float16* Vp = Vt + (size_t)kvh * HDm * Tn;

  if (lane < 16) { rowm[lane] = -1e30f; rowl[lane] = 0.f; }
  __syncthreads();

  v16h qa[4];
#pragma unroll
  for (int c = 0; c < 4; c++) {
    int* qi = (int*)&qa[c];
#pragma unroll
    for (int i = 0; i < 8; i++) {
      const int ko = (i < 4) ? (hi * 8 + 2 * i) : (16 + hi * 8 + 2 * (i - 4));
      qi[i] = *(const int*)(Qp + mr * HDm + c * 32 + ko);
    }
  }

  const v8f vzero = {};
  v8f oacc[8];
#pragma unroll
  for (int j = 0; j < 8; j++) oacc[j] = vzero;

  const float scl = 0.08838834764831845f;

  for (int k0 = 0; k0 < q0 + 16; k0 += 32) {
    __builtin_prefetch(Kp + (size_t)(k0 + 32 + mr) * HDm, 0, 1);
    __builtin_prefetch(Kp + (size_t)(k0 + 48 + mr) * HDm, 0, 1);

    v8f s0 = vzero, s1 = vzero;
#pragma unroll
    for (int c = 0; c < 4; c++) {
      v16h b0, b1;
      int* b0i = (int*)&b0;
      int* b1i = (int*)&b1;
      const _Float16* kb0 = Kp + (size_t)(k0 + mr) * HDm + c * 32;
      const _Float16* kb1 = kb0 + (size_t)16 * HDm;
#pragma unroll
      for (int i = 0; i < 8; i++) {
        const int ko = (i < 4) ? (hi * 8 + 2 * i) : (16 + hi * 8 + 2 * (i - 4));
        b0i[i] = *(const int*)(kb0 + ko);
        b1i[i] = *(const int*)(kb1 + ko);
      }
      s0 = __builtin_amdgcn_wmma_f32_16x16x32_f16(false, qa[c], false, b0, (short)0, s0, false, false);
      s1 = __builtin_amdgcn_wmma_f32_16x16x32_f16(false, qa[c], false, b1, (short)0, s1, false, false);
    }

    __syncthreads();

#pragma unroll
    for (int r = 0; r < 8; r++) {
      const int row = r + hi * 8;
      const int qrow = q0 + row;
      const float x0 = ((k0 + mr) <= qrow) ? s0[r] * scl : -1e30f;
      const float x1 = ((k0 + 16 + mr) <= qrow) ? s1[r] * scl : -1e30f;
      float tmax = fmaxf(x0, x1);
#pragma unroll
      for (int m = 8; m > 0; m >>= 1) tmax = fmaxf(tmax, __shfl_xor(tmax, m, 32));
      const float oldm = rowm[row];
      const float nm = fmaxf(oldm, tmax);
      const float alpha = expf(oldm - nm);
      const float p0 = expf(x0 - nm);
      const float p1 = expf(x1 - nm);
      float rs = p0 + p1;
#pragma unroll
      for (int m = 8; m > 0; m >>= 1) rs += __shfl_xor(rs, m, 32);
      if (mr == 0) { rowm[row] = nm; rowl[row] = rowl[row] * alpha + rs; }
      pbuf[row][mr] = (_Float16)p0;
      pbuf[row][16 + mr] = (_Float16)p1;
#pragma unroll
      for (int j = 0; j < 8; j++) oacc[j][r] *= alpha;
    }
    __syncthreads();

    v16h pa;
    int* pi = (int*)&pa;
#pragma unroll
    for (int i = 0; i < 8; i++) {
      const int ko = (i < 4) ? (hi * 8 + 2 * i) : (16 + hi * 8 + 2 * (i - 4));
      pi[i] = *(const int*)&pbuf[mr][ko];
    }
#pragma unroll
    for (int j = 0; j < 8; j++) {
      v16h bv;
      int* bvi = (int*)&bv;
      const _Float16* vb = Vp + (size_t)(j * 16 + mr) * Tn + k0;
#pragma unroll
      for (int i = 0; i < 8; i++) {
        const int ko = (i < 4) ? (hi * 8 + 2 * i) : (16 + hi * 8 + 2 * (i - 4));
        bvi[i] = *(const int*)(vb + ko);
      }
      oacc[j] = __builtin_amdgcn_wmma_f32_16x16x32_f16(false, pa, false, bv, (short)0, oacc[j], false, false);
    }
  }

  __syncthreads();
#pragma unroll
  for (int r = 0; r < 8; r++) {
    const int row = r + hi * 8;
    const float denom = fmaxf(rowl[row], 1e-20f);
#pragma unroll
    for (int j = 0; j < 8; j++) {
      Ao[(size_t)(q0 + row) * Hdim + (size_t)h * HDm + j * 16 + mr] = oacc[j][r] / denom;
    }
  }
}

__global__ void token_quant(const float* __restrict__ X, signed char* __restrict__ q,
                            float* __restrict__ s) {
  __shared__ float red[256];
  const int t = blockIdx.x;
  const float* row = X + (size_t)t * Hdim;
  float amax = 0.f;
  for (int i = threadIdx.x; i < Hdim; i += 256) amax = fmaxf(amax, fabsf(row[i]));
  red[threadIdx.x] = amax; __syncthreads();
  for (int st = 128; st > 0; st >>= 1) {
    if (threadIdx.x < st) red[threadIdx.x] = fmaxf(red[threadIdx.x], red[threadIdx.x + st]);
    __syncthreads();
  }
  const float scale = fmaxf(red[0] / 127.f, 1e-8f);
  if (threadIdx.x == 0) s[t] = scale;
  const float inv = 1.f / scale;
  for (int i = threadIdx.x; i < Hdim; i += 256) {
    int qq = (int)rintf(row[i] * inv);
    qq = qq < -127 ? -127 : (qq > 127 ? 127 : qq);
    q[(size_t)t * Hdim + i] = (signed char)qq;
  }
}

extern "C" void kernel_launch(void* const* d_in, const int* in_sizes, int n_in,
                              void* d_out, int out_size, void* d_ws, size_t ws_size,
                              hipStream_t stream) {
  (void)in_sizes; (void)n_in; (void)out_size; (void)ws_size;
  const float* x   = (const float*)d_in[0];
  const int*   pos = (const int*)d_in[1];
  const float* rw  = (const float*)d_in[2];
  const float* Wq  = (const float*)d_in[3];
  const float* Wk  = (const float*)d_in[4];
  const float* Wv  = (const float*)d_in[5];
  const float* Wo  = (const float*)d_in[6];

  size_t off = 0;
  auto take = [&](size_t bytes) -> void* {
    void* p = (char*)d_ws + off;
    off += (bytes + 255) & ~(size_t)255;
    return p;
  };
  signed char* xq  = (signed char*)take((size_t)Tn * Hdim);
  float*       sx  = (float*)take((size_t)Tn * 4);
  signed char* qWq = (signed char*)take((size_t)Hdim * Hdim);
  float*       sWq = (float*)take((size_t)Hdim * 4);
  signed char* qWk = (signed char*)take((size_t)KVD * Hdim);
  float*       sWk = (float*)take((size_t)KVD * 4);
  signed char* qWv = (signed char*)take((size_t)KVD * Hdim);
  float*       sWv = (float*)take((size_t)KVD * 4);
  signed char* qWo = (signed char*)take((size_t)Hdim * Hdim);
  float*       sWo = (float*)take((size_t)Hdim * 4);
  float* Yq = (float*)take((size_t)Tn * Hdim * 4);
  float* Yk = (float*)take((size_t)Tn * KVD * 4);
  float* Yv = (float*)take((size_t)Tn * KVD * 4);
  _Float16* Qh = (_Float16*)take((size_t)NHq * Tn * HDm * 2);
  _Float16* Kh = (_Float16*)take((size_t)NKVh * Tn * HDm * 2);
  _Float16* Vt = (_Float16*)take((size_t)NKVh * HDm * Tn * 2);
  float* Ao = (float*)take((size_t)Tn * Hdim * 4);
  signed char* aq = (signed char*)take((size_t)Tn * Hdim);
  float*       sa = (float*)take((size_t)Tn * 4);

  rmsnorm_quant<<<Tn, 256, 0, stream>>>(x, rw, xq, sx);
  weight_quant<<<Hdim, 256, 0, stream>>>(Wq, qWq, sWq);
  weight_quant<<<KVD,  256, 0, stream>>>(Wk, qWk, sWk);
  weight_quant<<<KVD,  256, 0, stream>>>(Wv, qWv, sWv);
  weight_quant<<<Hdim, 256, 0, stream>>>(Wo, qWo, sWo);

  gemm_i8_wmma<<<dim3(Hdim / 64, Tn / 32), 128, 0, stream>>>(xq, qWq, sx, sWq, Yq, Hdim);
  gemm_i8_wmma<<<dim3(KVD / 64,  Tn / 32), 128, 0, stream>>>(xq, qWk, sx, sWk, Yk, KVD);
  gemm_i8_wmma<<<dim3(KVD / 64,  Tn / 32), 128, 0, stream>>>(xq, qWv, sx, sWv, Yv, KVD);

  const int kvGroups = Tn * NKVh;
  quant4_group<<<kvGroups / 8, 256, 0, stream>>>(Yk, kvGroups);
  quant4_group<<<kvGroups / 8, 256, 0, stream>>>(Yv, kvGroups);

  rope_to_f16<<<dim3(Tn, NHq),  128, 0, stream>>>(Yq, pos, Qh, NHq);
  rope_to_f16<<<dim3(Tn, NKVh), 128, 0, stream>>>(Yk, pos, Kh, NKVh);
  v_to_f16_t<<<dim3(Tn, NKVh), 128, 0, stream>>>(Yv, Vt);

  attn_fmha<<<dim3(Tn / 16, NHq), 32, 0, stream>>>(Qh, Kh, Vt, Ao);

  token_quant<<<Tn, 256, 0, stream>>>(Ao, aq, sa);
  gemm_i8_wmma<<<dim3(Hdim / 64, Tn / 32), 128, 0, stream>>>(aq, qWo, sa, sWo,
                                                             (float*)d_out, Hdim);
}
